// LSTMModel_74732430950985
// MI455X (gfx1250) — matrix operand order repaired and checked by reading, not compiled, not run on hardware
//
#include <hip/hip_runtime.h>

typedef __attribute__((ext_vector_type(16))) _Float16 v16h;
typedef __attribute__((ext_vector_type(8)))  _Float16 v8h;
typedef __attribute__((ext_vector_type(8)))  float    v8f;
typedef __attribute__((ext_vector_type(4)))  float    v4f;

#define LSTM_T 512
#define LSTM_B 4096
#define TCH    64      // time steps staged in LDS per chunk

// ---- hardware tanh (CDNA5 V_TANH_F32), guarded, with exp2 fallback ---------
#if __has_builtin(__builtin_amdgcn_tanhf)
static __device__ __forceinline__ float htanh(float x) {
  return __builtin_amdgcn_tanhf(x);
}
#elif __has_builtin(__builtin_amdgcn_tanh_f32)
static __device__ __forceinline__ float htanh(float x) {
  return __builtin_amdgcn_tanh_f32(x);
}
#else
static __device__ __forceinline__ float htanh(float x) {
  float e = __builtin_amdgcn_exp2f(2.8853900817779268f * x);
  return (e - 1.0f) * __builtin_amdgcn_rcpf(e + 1.0f);
}
#endif

static __device__ __forceinline__ v8f wmma_f16(v16h a, v16h b, v8f c) {
  // D = A(16x32 f16) * B(32x16 f16) + C(16x16 f32)
  return __builtin_amdgcn_wmma_f32_16x16x32_f16(false, a, false, b, (short)0, c,
                                                false, false);
}

static __device__ __forceinline__ v8f vz8f() {
  v8f v;
#pragma unroll
  for (int i = 0; i < 8; ++i) v[i] = 0.0f;
  return v;
}
static __device__ __forceinline__ v16h vz16h() {
  v16h v;
#pragma unroll
  for (int i = 0; i < 16; ++i) v[i] = (_Float16)0.0f;
  return v;
}

// i,f,g,o gate tiles (16x16 f32, C layout) -> update c half, return h half.
// i/f/o pre-activations arrive PRE-SCALED by 0.5 (folded into weights/biases),
// so sigmoid(x) = 0.5 + 0.5*tanh(x/2) = fma(0.5, tanh(pre), 0.5).
static __device__ __forceinline__ v8f cellhalf(const v8f& gi, const v8f& gf,
                                               const v8f& gg, const v8f& go,
                                               v8f& c) {
  v8f h;
#pragma unroll
  for (int r = 0; r < 8; ++r) {
    float iv = 0.5f * htanh(gi[r]) + 0.5f;
    float fv = 0.5f * htanh(gf[r]) + 0.5f;
    float gv = htanh(gg[r]);                 // g gate: full-scale tanh
    float ov = 0.5f * htanh(go[r]) + 0.5f;
    float cv = fv * c[r] + iv * gv;
    c[r] = cv;
    h[r] = ov * htanh(cv);
  }
  return h;
}

// store h (C layout: lane=n, reg r -> row m=r+8*hi) as f16 row-major [16][32]
static __device__ __forceinline__ void store_h(_Float16* hb, const v8f& ha,
                                               const v8f& hbv, int n, int hi) {
#pragma unroll
  for (int r = 0; r < 8; ++r) {
    int m = r + 8 * hi;
    hb[m * 32 + n]      = (_Float16)ha[r];
    hb[m * 32 + 16 + n] = (_Float16)hbv[r];
  }
}

// load row-major [16][32] f16 buffer into WMMA A layout (16x32 f16):
// lane L (m = L&15, hi = L>>4): elems 0..15 = K 16*hi..16*hi+15
static __device__ __forceinline__ v16h load_A(const _Float16* hb, int m, int hi) {
  v8h lo = *(const v8h*)(hb + m * 32 + 16 * hi);
  v8h hv = *(const v8h*)(hb + m * 32 + 16 * hi + 8);
  v16h a;
#pragma unroll
  for (int e = 0; e < 8; ++e) {
    a[e]     = lo[e];
    a[e + 8] = hv[e];
  }
  return a;
}

__global__ __launch_bounds__(32)
void lstm2_wmma_kernel(const float* __restrict__ x,
                       const float* __restrict__ Wih0, const float* __restrict__ Whh0,
                       const float* __restrict__ bih0, const float* __restrict__ bhh0,
                       const float* __restrict__ Wih1, const float* __restrict__ Whh1,
                       const float* __restrict__ bih1, const float* __restrict__ bhh1,
                       const float* __restrict__ Wfc,  const float* __restrict__ bfc,
                       float* __restrict__ out) {
  __shared__ _Float16 hbuf[16 * 32];   // lane-transpose bounce buffer
  __shared__ float    xbuf[TCH * 16];  // staged x chunk [t][row]

  const int lane = threadIdx.x & 31;
  const int n    = lane & 15;  // B/C-layout column, also weight row mod 16
  const int hi   = lane >> 4;  // half-wave select
  const int b0   = blockIdx.x * 16;

  // ---- preload all weights into registers, WMMA B layout -------------------
  // B_t[k][n] = W[16*t + n][k], lane elem e -> k = e + 16*hi
  // Gate tiles: t=0,1 -> i ; 2,3 -> f ; 4,5 -> g ; 6,7 -> o.
  // i/f/o tiles (and matching biases / x-weights) pre-scaled by 0.5 so the
  // sigmoid is 0.5 + 0.5*tanh(pre) with no extra multiply in the hot loop.
  v16h wr0[8], wi1[8], wr1[8];
  float wx0[8], bb0[8], bb1[8];
#pragma unroll
  for (int t = 0; t < 8; ++t) {
    const float sc  = (t == 4 || t == 5) ? 1.0f : 0.5f;
    const int   row = 16 * t + n;
    const float4* p0 = (const float4*)(Whh0 + row * 32 + 16 * hi);
    const float4* p1 = (const float4*)(Wih1 + row * 32 + 16 * hi);
    const float4* p2 = (const float4*)(Whh1 + row * 32 + 16 * hi);
#pragma unroll
    for (int q = 0; q < 4; ++q) {
      float4 a0 = p0[q], a1 = p1[q], a2 = p2[q];
      wr0[t][4 * q + 0] = (_Float16)(sc * a0.x);
      wr0[t][4 * q + 1] = (_Float16)(sc * a0.y);
      wr0[t][4 * q + 2] = (_Float16)(sc * a0.z);
      wr0[t][4 * q + 3] = (_Float16)(sc * a0.w);
      wi1[t][4 * q + 0] = (_Float16)(sc * a1.x);
      wi1[t][4 * q + 1] = (_Float16)(sc * a1.y);
      wi1[t][4 * q + 2] = (_Float16)(sc * a1.z);
      wi1[t][4 * q + 3] = (_Float16)(sc * a1.w);
      wr1[t][4 * q + 0] = (_Float16)(sc * a2.x);
      wr1[t][4 * q + 1] = (_Float16)(sc * a2.y);
      wr1[t][4 * q + 2] = (_Float16)(sc * a2.z);
      wr1[t][4 * q + 3] = (_Float16)(sc * a2.w);
    }
    wx0[t] = sc * Wih0[row];                      // I == 1
    bb0[t] = sc * (bih0[row] + bhh0[row]);
    bb1[t] = sc * (bih1[row] + bhh1[row]);
  }

  // ---- states (zero-initialized) -------------------------------------------
  v8f c0a = vz8f(), c0b = vz8f(), c1a = vz8f(), c1b = vz8f();
  v16h Ah0 = vz16h(), Ah1 = vz16h();
  v8f h1a = vz8f(), h1b = vz8f();

  // ---- fused 2-layer time loop ---------------------------------------------
  for (int tc = 0; tc < LSTM_T / TCH; ++tc) {
    // stage x chunk: lane (r = n, half = hi) copies 32 contiguous steps of row r
    {
      const float* xg = x + (size_t)(b0 + n) * LSTM_T + tc * TCH + 32 * hi;
      if (tc + 1 < LSTM_T / TCH) __builtin_prefetch(xg + TCH, 0, 1);
      const float4* xg4 = (const float4*)xg;
#pragma unroll
      for (int q = 0; q < 8; ++q) {
        float4 v = xg4[q];
        int tt = 32 * hi + 4 * q;
        xbuf[(tt + 0) * 16 + n] = v.x;
        xbuf[(tt + 1) * 16 + n] = v.y;
        xbuf[(tt + 2) * 16 + n] = v.z;
        xbuf[(tt + 3) * 16 + n] = v.w;
      }
    }

#pragma unroll 1
    for (int tl = 0; tl < TCH; ++tl) {
      // x values for this lane's 8 rows (m = r + 8*hi)
      v4f xa = *(const v4f*)(xbuf + tl * 16 + 8 * hi);
      v4f xb = *(const v4f*)(xbuf + tl * 16 + 8 * hi + 4);
      v8f x8;
#pragma unroll
      for (int r = 0; r < 4; ++r) { x8[r] = xa[r]; x8[r + 4] = xb[r]; }

      // ---------------- layer 0: gates = x*Wih0 + b + h0 @ Whh0^T -----------
      v8f h0a, h0b;
#pragma unroll
      for (int kk = 0; kk < 2; ++kk) {
        v8f gi, gf, gg, go;
#pragma unroll
        for (int r = 0; r < 8; ++r) {
          gi[r] = bb0[0 + kk] + x8[r] * wx0[0 + kk];
          gf[r] = bb0[2 + kk] + x8[r] * wx0[2 + kk];
          gg[r] = bb0[4 + kk] + x8[r] * wx0[4 + kk];
          go[r] = bb0[6 + kk] + x8[r] * wx0[6 + kk];
        }
        gi = wmma_f16(Ah0, wr0[0 + kk], gi);
        gf = wmma_f16(Ah0, wr0[2 + kk], gf);
        gg = wmma_f16(Ah0, wr0[4 + kk], gg);
        go = wmma_f16(Ah0, wr0[6 + kk], go);
        v8f& c = kk ? c0b : c0a;
        v8f h = cellhalf(gi, gf, gg, go, c);
        if (kk == 0) h0a = h; else h0b = h;
      }
      store_h(hbuf, h0a, h0b, n, hi);     // cross-lane transpose via LDS
      Ah0 = load_A(hbuf, n, hi);          // h0_t in A layout (f16)

      // ---------------- layer 1: gates = h0 @ Wih1^T + b + h1 @ Whh1^T ------
#pragma unroll
      for (int kk = 0; kk < 2; ++kk) {
        v8f gi, gf, gg, go;
#pragma unroll
        for (int r = 0; r < 8; ++r) {
          gi[r] = bb1[0 + kk];
          gf[r] = bb1[2 + kk];
          gg[r] = bb1[4 + kk];
          go[r] = bb1[6 + kk];
        }
        gi = wmma_f16(Ah0, wi1[0 + kk], gi);
        gf = wmma_f16(Ah0, wi1[2 + kk], gf);
        gg = wmma_f16(Ah0, wi1[4 + kk], gg);
        go = wmma_f16(Ah0, wi1[6 + kk], go);
        gi = wmma_f16(Ah1, wr1[0 + kk], gi);
        gf = wmma_f16(Ah1, wr1[2 + kk], gf);
        gg = wmma_f16(Ah1, wr1[4 + kk], gg);
        go = wmma_f16(Ah1, wr1[6 + kk], go);
        v8f& c = kk ? c1b : c1a;
        v8f h = cellhalf(gi, gf, gg, go, c);
        if (kk == 0) h1a = h; else h1b = h;
      }
      store_h(hbuf, h1a, h1b, n, hi);
      Ah1 = load_A(hbuf, n, hi);          // h1_t in A layout (f16)
    }
  }

  // ---- final FC on h1[:, T-1, :]: out = relu(h1 @ Wfc^T + bfc) -------------
  // write last h1 (f32, C layout) row-major into xbuf (reused as [16][32] f32)
#pragma unroll
  for (int r = 0; r < 8; ++r) {
    int m = r + 8 * hi;
    xbuf[m * 32 + n]      = h1a[r];
    xbuf[m * 32 + 16 + n] = h1b[r];
  }
  __builtin_amdgcn_wave_barrier();
  if (lane < 16) {
    float acc = bfc[0];
    const v4f* wp = (const v4f*)Wfc;
    const v4f* hp = (const v4f*)(xbuf + lane * 32);
#pragma unroll
    for (int q = 0; q < 8; ++q) {
      v4f w = wp[q], h = hp[q];
      acc += w[0] * h[0] + w[1] * h[1] + w[2] * h[2] + w[3] * h[3];
    }
    out[b0 + lane] = acc > 0.0f ? acc : 0.0f;
  }
}

extern "C" void kernel_launch(void* const* d_in, const int* in_sizes, int n_in,
                              void* d_out, int out_size, void* d_ws, size_t ws_size,
                              hipStream_t stream) {
  (void)in_sizes; (void)n_in; (void)out_size; (void)d_ws; (void)ws_size;
  const float* x    = (const float*)d_in[0];
  const float* Wih0 = (const float*)d_in[1];
  const float* Whh0 = (const float*)d_in[2];
  const float* bih0 = (const float*)d_in[3];
  const float* bhh0 = (const float*)d_in[4];
  const float* Wih1 = (const float*)d_in[5];
  const float* Whh1 = (const float*)d_in[6];
  const float* bih1 = (const float*)d_in[7];
  const float* bhh1 = (const float*)d_in[8];
  const float* Wfc  = (const float*)d_in[9];
  const float* bfc  = (const float*)d_in[10];
  float* out = (float*)d_out;

  // one wave32 per 16 batch rows: 4096/16 = 256 single-wave workgroups
  lstm2_wmma_kernel<<<LSTM_B / 16, 32, 0, stream>>>(
      x, Wih0, Whh0, bih0, bhh0, Wih1, Whh1, bih1, bhh1, Wfc, bfc, out);
}
